// FAMHA_69664369541164
// MI455X (gfx1250) — matrix operand order repaired and checked by reading, not compiled, not run on hardware
//
#include <hip/hip_runtime.h>

// ---- problem constants ----
#define B_SZ   64
#define NQ     784
#define NK     196
#define NKP    224          // nk padded to multiple of 32
#define C_DIM  512
#define HHEADS 8
#define DK     64
#define LN_EPS 1e-5f

typedef _Float16 v16h __attribute__((ext_vector_type(16)));
typedef _Float16 v8h  __attribute__((ext_vector_type(8)));
typedef float    v8f  __attribute__((ext_vector_type(8)));
typedef float    v4f  __attribute__((ext_vector_type(4)));

__device__ __forceinline__ v8f zero8() {
  v8f z = {0.f,0.f,0.f,0.f,0.f,0.f,0.f,0.f};
  return z;
}

#define WMMA_F16(a,b,c) \
  __builtin_amdgcn_wmma_f32_16x16x32_f16(false,(a),false,(b),(short)0,(c),false,false)

// A-fragment (16x32 f16): lane = row (L%16); halves K = base+0..7 and base+16..23, base=(L/16)*8
__device__ __forceinline__ v16h frag_a_f16(const _Float16* A, int lda) {
  const int l = threadIdx.x & 31;
  const _Float16* p = A + (l & 15) * lda + ((l >> 4) << 3);
  v8h lo = *(const v8h*)(p);
  v8h hi = *(const v8h*)(p + 16);
  v16h r;
#pragma unroll
  for (int i = 0; i < 8; ++i) { r[i] = lo[i]; r[i + 8] = hi[i]; }
  return r;
}

// A-fragment sourced from fp32 (converted inline)
__device__ __forceinline__ v16h frag_a_f32(const float* A, int lda) {
  const int l = threadIdx.x & 31;
  const float* p = A + (l & 15) * lda + ((l >> 4) << 3);
  v4f a0 = *(const v4f*)(p);
  v4f a1 = *(const v4f*)(p + 4);
  v4f b0 = *(const v4f*)(p + 16);
  v4f b1 = *(const v4f*)(p + 20);
  v16h r;
#pragma unroll
  for (int i = 0; i < 4; ++i) {
    r[i]      = (_Float16)a0[i];
    r[i + 4]  = (_Float16)a1[i];
    r[i + 8]  = (_Float16)b0[i];
    r[i + 12] = (_Float16)b1[i];
  }
  return r;
}

// B-fragment (32x16 f16): lane = col (L%16); halves K = base+0..7 and base+16..23, base=(L/16)*8
__device__ __forceinline__ v16h frag_b_f16(const _Float16* B, int ldb) {
  const int l = threadIdx.x & 31;
  const _Float16* p = B + (l & 15) * ldb + ((l >> 4) << 3);
  v8h lo = *(const v8h*)(p);
  v8h hi = *(const v8h*)(p + 16);
  v16h r;
#pragma unroll
  for (int i = 0; i < 8; ++i) { r[i] = lo[i]; r[i + 8] = hi[i]; }
  return r;
}

// ---------------------------------------------------------------------------
__global__ void cvt_f32_to_f16(const float* __restrict__ src,
                               _Float16* __restrict__ dst, int n) {
  int i = blockIdx.x * blockDim.x + threadIdx.x;
  if (i < n) dst[i] = (_Float16)src[i];
}

// ---------------------------------------------------------------------------
// Per (batch, 16-key-row tile): spatial reduction + affine + LayerNorm -> LDS,
// then K = x*Wk^T+bk  (f16 [b,224,512]) and V^T (f16 [b,h,64,224]).
__global__ void kv_kernel(const float* __restrict__ queries,
                          const _Float16* __restrict__ Wk_h, const float* __restrict__ bk,
                          const _Float16* __restrict__ Wv_h, const float* __restrict__ bv,
                          const float* __restrict__ sr_w, const float* __restrict__ sr_b,
                          const float* __restrict__ ln_w, const float* __restrict__ ln_b,
                          _Float16* __restrict__ Kh, _Float16* __restrict__ Vth) {
  __shared__ _Float16 xt[16 * C_DIM];

  const int b  = blockIdx.x / (NKP / 16);
  const int kt = blockIdx.x % (NKP / 16);
  const int w  = threadIdx.x >> 5;
  const int l  = threadIdx.x & 31;

  // ---- build 16 layer-normed rows (2 rows per wave) ----
#pragma unroll
  for (int rr = 0; rr < 2; ++rr) {
    const int r  = w * 2 + rr;
    int kj = kt * 16 + r;
    if (kj >= NK) kj = NK - 1;          // pad rows: finite dup data, masked later
    const int i2 = kj / 14, j2 = kj % 14;
    const int pos = 56 * i2 + 2 * j2;   // stride-2 subsample of 28x28
    const float* src = queries + ((size_t)b * NQ + pos) * C_DIM;

    float vals[16];
    float s = 0.f, s2 = 0.f;
#pragma unroll
    for (int t = 0; t < 16; ++t) {
      const int c = l + 32 * t;
      const float v = src[c] * sr_w[c] + sr_b[c];
      vals[t] = v; s += v; s2 += v * v;
    }
#pragma unroll
    for (int o = 16; o >= 1; o >>= 1) {
      s  += __shfl_xor(s,  o, 32);
      s2 += __shfl_xor(s2, o, 32);
    }
    const float mu   = s * (1.0f / C_DIM);
    const float var  = s2 * (1.0f / C_DIM) - mu * mu;
    const float rstd = rsqrtf(var + LN_EPS);
#pragma unroll
    for (int t = 0; t < 16; ++t) {
      const int c = l + 32 * t;
      xt[r * C_DIM + c] = (_Float16)((vals[t] - mu) * rstd * ln_w[c] + ln_b[c]);
    }
  }
  __syncthreads();

  // ---- GEMM: 64 jobs = {K,V} x 32 n-tiles, 8 per wave ----
  const int rowM = (l >> 4) << 3;
  const int colN = l & 15;
#pragma unroll 1
  for (int j = 0; j < 8; ++j) {
    const int job = w + 8 * j;
    const int mat = job >> 5;           // 0 = K, 1 = V
    const int nt  = job & 31;
    const _Float16* W    = mat ? Wv_h : Wk_h;
    const float*    bias = mat ? bv   : bk;
    v8f acc = zero8();
#pragma unroll 1
    for (int ks = 0; ks < 16; ++ks) {
      v16h a  = frag_a_f16(xt + ks * 32, C_DIM);
      v16h bf = frag_b_f16(W + (size_t)(nt * 16) * C_DIM + ks * 32, C_DIM);
      acc = WMMA_F16(a, bf, acc);
    }
    const int o = nt * 16 + colN;
    const float bb = bias[o];
    if (mat == 0) {
#pragma unroll
      for (int i = 0; i < 8; ++i) {
        const int kr = kt * 16 + rowM + i;
        Kh[((size_t)b * NKP + kr) * C_DIM + o] = (_Float16)(acc[i] + bb);
      }
    } else {
      const int h = o >> 6, d = o & 63;
#pragma unroll
      for (int i = 0; i < 8; ++i) {
        const int kr = kt * 16 + rowM + i;
        Vth[(((size_t)b * HHEADS + h) * DK + d) * NKP + kr] = (_Float16)(acc[i] + bb);
      }
    }
  }
}

// ---------------------------------------------------------------------------
// Fully fused attention per (batch, 16-query tile). 8 waves = 8 heads.
// Dynamic LDS: OT half[16][512] (16KB, Q-tile then out-tile) + S half[8][16][224] (56KB).
__global__ void attn_kernel(const float* __restrict__ queries,
                            const _Float16* __restrict__ Wq_h, const float* __restrict__ bq,
                            const _Float16* __restrict__ Wo_h, const float* __restrict__ bo,
                            const _Float16* __restrict__ Kh, const _Float16* __restrict__ Vth,
                            const float* __restrict__ tw, const float* __restrict__ tb,
                            float* __restrict__ out) {
  extern __shared__ __align__(16) char smem[];
  _Float16* OT = (_Float16*)smem;                          // [16][512]
  _Float16* S  = (_Float16*)(smem + 16 * C_DIM * 2);       // [8][16][NKP]

  const int b  = blockIdx.x / (NQ / 16);
  const int q0 = (blockIdx.x % (NQ / 16)) * 16;
  const int w  = threadIdx.x >> 5;
  const int l  = threadIdx.x & 31;
  const int rowM = (l >> 4) << 3;
  const int colN = l & 15;

  // ---- Phase 1: Q tile = queries * Wq^T + bq -> OT (f16) ----
  {
    v8f acc[4] = {zero8(), zero8(), zero8(), zero8()};
#pragma unroll 1
    for (int ks = 0; ks < 16; ++ks) {
      v16h a = frag_a_f32(queries + ((size_t)b * NQ + q0) * C_DIM + ks * 32, C_DIM);
#pragma unroll
      for (int j = 0; j < 4; ++j) {
        const int nt = w + 8 * j;
        v16h bf = frag_b_f16(Wq_h + (size_t)(nt * 16) * C_DIM + ks * 32, C_DIM);
        acc[j] = WMMA_F16(a, bf, acc[j]);
      }
    }
#pragma unroll
    for (int j = 0; j < 4; ++j) {
      const int nt = w + 8 * j;
      const float bb = bq[nt * 16 + colN];
#pragma unroll
      for (int i = 0; i < 8; ++i)
        OT[(rowM + i) * C_DIM + nt * 16 + colN] = (_Float16)(acc[j][i] + bb);
    }
  }
  __syncthreads();

  // ---- Phase 2: scores S[h] = (Q_h K_h^T) / 8, wave w = head w ----
  {
    const int h = w;
    v16h a0 = frag_a_f16(OT + h * DK, C_DIM);
    v16h a1 = frag_a_f16(OT + h * DK + 32, C_DIM);
#pragma unroll 1
    for (int ktile = 0; ktile < NKP / 16; ++ktile) {
      const _Float16* Kbase = Kh + ((size_t)b * NKP + ktile * 16) * C_DIM + h * DK;
      v16h b0 = frag_b_f16(Kbase, C_DIM);
      v16h b1 = frag_b_f16(Kbase + 32, C_DIM);
      v8f acc = zero8();
      acc = WMMA_F16(a0, b0, acc);
      acc = WMMA_F16(a1, b1, acc);
#pragma unroll
      for (int i = 0; i < 8; ++i)
        S[(h * 16 + rowM + i) * NKP + ktile * 16 + colN] = (_Float16)(acc[i] * 0.125f);
    }
  }
  __syncthreads();

  // ---- Phase 3a: 1x1 conv across heads, in place (each (q,k) owned by one thread) ----
  {
    float twr[64], tbr[8];
#pragma unroll
    for (int i = 0; i < 64; ++i) twr[i] = tw[i];
#pragma unroll
    for (int i = 0; i < 8; ++i) tbr[i] = tb[i];
    for (int p = threadIdx.x; p < 16 * NKP; p += 256) {
      const int q = p / NKP, k = p % NKP;
      float sv[8];
#pragma unroll
      for (int hh = 0; hh < 8; ++hh) sv[hh] = (float)S[(hh * 16 + q) * NKP + k];
#pragma unroll
      for (int g = 0; g < 8; ++g) {
        float m = tbr[g];
#pragma unroll
        for (int hh = 0; hh < 8; ++hh) m += twr[g * 8 + hh] * sv[hh];
        S[(g * 16 + q) * NKP + k] = (_Float16)m;
      }
    }
  }
  __syncthreads();

  // ---- Phase 3b: masked softmax over k (pad cols -> 0), wave w = head w ----
  {
    const int h = w;
#pragma unroll 1
    for (int q = 0; q < 16; ++q) {
      _Float16* row = S + (h * 16 + q) * NKP;
      float v[7];
      float mx = -1e30f;
#pragma unroll
      for (int t = 0; t < 7; ++t) {
        const int k = l + 32 * t;
        const float x = (k < NK) ? (float)row[k] : -1e30f;
        v[t] = x;
        mx = fmaxf(mx, x);
      }
#pragma unroll
      for (int o = 16; o >= 1; o >>= 1) mx = fmaxf(mx, __shfl_xor(mx, o, 32));
      float sum = 0.f;
#pragma unroll
      for (int t = 0; t < 7; ++t) {
        const int k = l + 32 * t;
        const float e = (k < NK) ? __expf(v[t] - mx) : 0.f;
        v[t] = e; sum += e;
      }
#pragma unroll
      for (int o = 16; o >= 1; o >>= 1) sum += __shfl_xor(sum, o, 32);
      const float inv = 1.0f / sum;
#pragma unroll
      for (int t = 0; t < 7; ++t) row[l + 32 * t] = (_Float16)(v[t] * inv);
    }
  }
  __syncthreads();

  // ---- Phase 4: att @ V -> OT (Q tile is dead), wave w = head w ----
  {
    const int h = w;
#pragma unroll 1
    for (int nt = 0; nt < 4; ++nt) {
      v8f acc = zero8();
#pragma unroll 1
      for (int ks = 0; ks < NKP / 32; ++ks) {
        v16h a  = frag_a_f16(S + (size_t)(h * 16) * NKP + ks * 32, NKP);
        v16h bf = frag_b_f16(Vth + (((size_t)b * HHEADS + h) * DK + nt * 16) * NKP + ks * 32, NKP);
        acc = WMMA_F16(a, bf, acc);
      }
#pragma unroll
      for (int i = 0; i < 8; ++i)
        OT[(rowM + i) * C_DIM + h * DK + nt * 16 + colN] = (_Float16)acc[i];
    }
  }
  __syncthreads();

  // ---- Phase 5: out = OT * Wo^T + bo -> fp32 ----
  {
    v8f acc[4] = {zero8(), zero8(), zero8(), zero8()};
#pragma unroll 1
    for (int ks = 0; ks < 16; ++ks) {
      v16h a = frag_a_f16(OT + ks * 32, C_DIM);
#pragma unroll
      for (int j = 0; j < 4; ++j) {
        const int nt = w + 8 * j;
        v16h bf = frag_b_f16(Wo_h + (size_t)(nt * 16) * C_DIM + ks * 32, C_DIM);
        acc[j] = WMMA_F16(a, bf, acc[j]);
      }
    }
#pragma unroll
    for (int j = 0; j < 4; ++j) {
      const int nt = w + 8 * j;
      const float bb = bo[nt * 16 + colN];
#pragma unroll
      for (int i = 0; i < 8; ++i)
        out[((size_t)b * NQ + q0 + rowM + i) * C_DIM + nt * 16 + colN] = acc[j][i] + bb;
    }
  }
}

// ---------------------------------------------------------------------------
extern "C" void kernel_launch(void* const* d_in, const int* in_sizes, int n_in,
                              void* d_out, int out_size, void* d_ws, size_t ws_size,
                              hipStream_t stream) {
  const float* queries = (const float*)d_in[0];
  // d_in[1]=keys, d_in[2]=values are unused by the reference
  const float* Wq   = (const float*)d_in[3];
  const float* bq   = (const float*)d_in[4];
  const float* Wk   = (const float*)d_in[5];
  const float* bk   = (const float*)d_in[6];
  const float* Wv   = (const float*)d_in[7];
  const float* bv   = (const float*)d_in[8];
  const float* Wo   = (const float*)d_in[9];
  const float* bo   = (const float*)d_in[10];
  const float* sr_w = (const float*)d_in[11];
  const float* sr_b = (const float*)d_in[12];
  const float* ln_w = (const float*)d_in[13];
  const float* ln_b = (const float*)d_in[14];
  const float* tw   = (const float*)d_in[15];
  const float* tb   = (const float*)d_in[16];
  float* out = (float*)d_out;

  _Float16* ws = (_Float16*)d_ws;
  const size_t WSZ = (size_t)C_DIM * C_DIM;
  _Float16* Wq_h = ws;
  _Float16* Wk_h = Wq_h + WSZ;
  _Float16* Wv_h = Wk_h + WSZ;
  _Float16* Wo_h = Wv_h + WSZ;
  _Float16* Kh   = Wo_h + WSZ;                              // [64][224][512]
  _Float16* Vth  = Kh + (size_t)B_SZ * NKP * C_DIM;         // [64][8][64][224]

  const int n = C_DIM * C_DIM;
  cvt_f32_to_f16<<<(n + 255) / 256, 256, 0, stream>>>(Wq, Wq_h, n);
  cvt_f32_to_f16<<<(n + 255) / 256, 256, 0, stream>>>(Wk, Wk_h, n);
  cvt_f32_to_f16<<<(n + 255) / 256, 256, 0, stream>>>(Wv, Wv_h, n);
  cvt_f32_to_f16<<<(n + 255) / 256, 256, 0, stream>>>(Wo, Wo_h, n);

  kv_kernel<<<B_SZ * (NKP / 16), 256, 0, stream>>>(
      queries, Wk_h, bk, Wv_h, bv, sr_w, sr_b, ln_w, ln_b, Kh, Vth);

  const size_t shmem = (size_t)16 * C_DIM * 2 + (size_t)HHEADS * 16 * NKP * 2;  // 73728 B
  attn_kernel<<<B_SZ * (NQ / 16), 256, shmem, stream>>>(
      queries, Wq_h, bq, Wo_h, bo, Kh, Vth, tw, tb, out);
}
